// ManyDeepFeatureMatcher_52785148067940
// MI455X (gfx1250) — matrix operand order repaired and checked by reading, not compiled, not run on hardware
//
#include <hip/hip_runtime.h>
#include <hip/hip_bf16.h>

// Problem constants (from reference): D=704, N1=N2=8192, RATIO=1.0, EPS=1e-8
#define D_DIM   704
#define N_PTS   8192
#define NSEG    8
#define SEG_COLS (N_PTS / NSEG)   // 1024
#define BM      128
#define BN      128
#define BK      32
#define K_EPS   1e-8f
#define K_RATIO 1.0f

#define LDS_PLANE 10240           // 128 cols * 80B stride
#define LDS_BUF   (2 * LDS_PLANE) // hi + lo planes
#define KSTEPS    (D_DIM / BK)    // 22
#define NTILES    (SEG_COLS / BN) // 8
#define NSTEPS    (NTILES * KSTEPS)

typedef __attribute__((ext_vector_type(16))) __bf16 bf16x16;
typedef __attribute__((ext_vector_type(8)))  __bf16 bf16x8;
typedef __attribute__((ext_vector_type(8)))  float  f32x8;

// ---------------------------------------------------------------------------
// Pass 1: per-column L2 norm of a (D_DIM x N_PTS) row-major matrix.
// ---------------------------------------------------------------------------
__global__ void colnorm_kernel(const float* __restrict__ src, float* __restrict__ rn) {
    int col = blockIdx.x * blockDim.x + threadIdx.x;
    if (col >= N_PTS) return;
    float s = 0.f;
    for (int k = 0; k < D_DIM; ++k) {
        float v = src[(size_t)k * N_PTS + col];
        s += v * v;
    }
    rn[col] = rsqrtf(s);
}

// ---------------------------------------------------------------------------
// Pass 2: transpose (D x N) -> (N x D), normalize, split fp32 into bf16 hi +
// bf16 residual lo planes (K-contiguous rows, ready for WMMA fragment loads).
// ---------------------------------------------------------------------------
__global__ void xpose_split_kernel(const float* __restrict__ src, const float* __restrict__ rn,
                                   __bf16* __restrict__ hi, __bf16* __restrict__ lo) {
    __shared__ float tile[32 * 33];
    const int tx = threadIdx.x & 31;
    const int ty = threadIdx.x >> 5;
    const int i0 = blockIdx.x * 32;
    const int k0 = blockIdx.y * 32;
    const float scale = rn[i0 + tx];
    #pragma unroll
    for (int kk = 0; kk < 4; ++kk) {
        int kl = ty + kk * 8;
        tile[tx * 33 + kl] = src[(size_t)(k0 + kl) * N_PTS + (i0 + tx)] * scale;
    }
    __syncthreads();
    #pragma unroll
    for (int kk = 0; kk < 4; ++kk) {
        int il = ty + kk * 8;
        float v = tile[il * 33 + tx];
        __bf16 h = (__bf16)v;
        __bf16 l = (__bf16)(v - (float)h);
        size_t o = (size_t)(i0 + il) * D_DIM + (k0 + tx);
        hi[o] = h;
        lo[o] = l;
    }
}

// top-2 insert (second-best index never needed downstream)
__device__ __forceinline__ void top2_ins(float& v0, int& i0, float& v1, float v, int i) {
    if (v > v0) { v1 = v0; v0 = v; i0 = i; }
    else if (v > v1) { v1 = v; }
}

// ---------------------------------------------------------------------------
// Fused GEMM + running row-top2, software-pipelined with CDNA5 async
// global->LDS copies (ASYNCcnt) and double-buffered LDS B tiles.
// Grid: (N_PTS/BM, NSEG). Block: 256 threads = 8 waves; wave w owns a
// 16-row strip. Per 16x16x32 step: 3 bf16 WMMAs (hi*hi, hi*lo, lo*hi).
// ---------------------------------------------------------------------------
__global__ __launch_bounds__(256)
void simtop2_kernel(const __bf16* __restrict__ Xhi, const __bf16* __restrict__ Xlo,
                    const __bf16* __restrict__ Yhi, const __bf16* __restrict__ Ylo,
                    float4* __restrict__ part) {
    __shared__ __align__(16) char smem[2 * LDS_BUF];   // double-buffered B tile
    const int lane = threadIdx.x & 31;
    const int wave = threadIdx.x >> 5;
    const int M0   = blockIdx.x * BM + wave * 16;
    const int seg  = blockIdx.y;

    const int lrow = lane & 15;
    const int aoff = (lane < 16) ? 0 : 8;   // A-frag K sub-offset (ISA layout)
    const int boff = (lane < 16) ? 0 : 8;   // B-frag K sub-offset (ISA layout)

    // Per-thread staging role: one (plane, column) = 64 contiguous bytes.
    const int st_plane = threadIdx.x >> 7;
    const int st_col   = threadIdx.x & 127;
    const __bf16* st_src = (st_plane ? Ylo : Yhi);
    const unsigned st_lds_base =
        (unsigned)(size_t)(smem + st_plane * LDS_PLANE + st_col * 80);

    // Issue one async B-tile stage: 64B per thread via 4x async b128
    // (same INST_OFFSET advances both global src and LDS dst).
    auto stage = [&](int st) {
        const int n0 = seg * SEG_COLS + (st / KSTEPS) * BN;
        const int kk = (st % KSTEPS) * BK;
        unsigned long long g =
            (unsigned long long)(size_t)(st_src + (size_t)(n0 + st_col) * D_DIM + kk);
        unsigned lds = st_lds_base + (unsigned)((st & 1) * LDS_BUF);
        asm volatile(
            "global_load_async_to_lds_b128 %0, %1, off\n\t"
            "global_load_async_to_lds_b128 %0, %1, off offset:16\n\t"
            "global_load_async_to_lds_b128 %0, %1, off offset:32\n\t"
            "global_load_async_to_lds_b128 %0, %1, off offset:48"
            :: "v"(lds), "v"(g) : "memory");
    };

    float rv0[8], rv1[8];
    int   ri0[8];
    #pragma unroll
    for (int r = 0; r < 8; ++r) { rv0[r] = -1e30f; rv1[r] = -1e30f; ri0[r] = -1; }

    f32x8 acc[8];
    int kt = 0;                       // k-step within current tile
    int N0 = seg * SEG_COLS;          // current tile column base

    stage(0);                         // prologue

    for (int st = 0; st < NSTEPS; ++st) {
        // Current step's buffer is complete once our asyncs retired and all
        // waves arrive; reads of the previous buffer were register-consumed
        // (s_wait_dscnt before WMMA) before each wave reaches this barrier.
        asm volatile("s_wait_asynccnt 0x0" ::: "memory");
        __syncthreads();

        if (st + 1 < NSTEPS) stage(st + 1);   // overlap next copy with math

        if (kt == 0) {
            #pragma unroll
            for (int s = 0; s < 8; ++s)
                acc[s] = (f32x8){0.f,0.f,0.f,0.f,0.f,0.f,0.f,0.f};
        }
        const int kk = kt * BK;
        const char* buf = smem + (st & 1) * LDS_BUF;

        // A fragments (16 rows x 32 K) hi & lo straight from global (L2-resident)
        bf16x16 ah, al;
        {
            const size_t base = (size_t)(M0 + lrow) * D_DIM + kk + aoff;
            bf16x8 p0 = *(const bf16x8*)(Xhi + base);
            bf16x8 p1 = *(const bf16x8*)(Xhi + base + 16);
            bf16x8 q0 = *(const bf16x8*)(Xlo + base);
            bf16x8 q1 = *(const bf16x8*)(Xlo + base + 16);
            #pragma unroll
            for (int i = 0; i < 8; ++i) {
                ah[i] = p0[i]; ah[i + 8] = p1[i];
                al[i] = q0[i]; al[i + 8] = q1[i];
            }
        }

        #pragma unroll
        for (int s = 0; s < 8; ++s) {
            const char* bb = buf + (s * 16 + lrow) * 80 + boff * 2;
            bf16x8 b0h = *(const bf16x8*)(bb);
            bf16x8 b1h = *(const bf16x8*)(bb + 32);
            bf16x8 b0l = *(const bf16x8*)(bb + LDS_PLANE);
            bf16x8 b1l = *(const bf16x8*)(bb + LDS_PLANE + 32);
            bf16x16 bh, bl;
            #pragma unroll
            for (int i = 0; i < 8; ++i) {
                bh[i] = b0h[i]; bh[i + 8] = b1h[i];
                bl[i] = b0l[i]; bl[i + 8] = b1l[i];
            }
            acc[s] = __builtin_amdgcn_wmma_f32_16x16x32_bf16(false, ah, false, bh, (short)0, acc[s], false, false);
            acc[s] = __builtin_amdgcn_wmma_f32_16x16x32_bf16(false, ah, false, bl, (short)0, acc[s], false, false);
            acc[s] = __builtin_amdgcn_wmma_f32_16x16x32_bf16(false, al, false, bh, (short)0, acc[s], false, false);
        }

        if (++kt == KSTEPS) {
            kt = 0;
            // Tile done: top-2 across its 128 columns.
            // Acc element e of lane L is row (e + 8*(L/16)), col (s*16 + L%16):
            // reduce per-lane over s, then xor over lane bits 0..3.
            #pragma unroll
            for (int r = 0; r < 8; ++r) {
                float v0 = -1e30f, v1 = -1e30f; int i0 = -1;
                #pragma unroll
                for (int s = 0; s < 8; ++s)
                    top2_ins(v0, i0, v1, acc[s][r], N0 + s * 16 + lrow);
                #pragma unroll
                for (int m = 1; m < 16; m <<= 1) {
                    float ov0 = __shfl_xor(v0, m, 32);
                    float ov1 = __shfl_xor(v1, m, 32);
                    int   oi0 = __shfl_xor(i0, m, 32);
                    top2_ins(v0, i0, v1, ov0, oi0);
                    top2_ins(v0, i0, v1, ov1, -2);
                }
                top2_ins(rv0[r], ri0[r], rv1[r], v0, i0);
                top2_ins(rv0[r], ri0[r], rv1[r], v1, -2);
            }
            N0 += BN;
        }
    }

    // lane 0 -> rows M0..M0+7, lane 16 -> rows M0+8..M0+15
    if (lrow == 0) {
        const int half = lane >> 4;
        #pragma unroll
        for (int r = 0; r < 8; ++r) {
            const int row = M0 + half * 8 + r;
            part[(size_t)seg * N_PTS + row] =
                make_float4(rv0[r], rv1[r], __int_as_float(ri0[r]), 0.f);
        }
    }
}

// ---------------------------------------------------------------------------
// Merge NSEG partial top-2s per row.
// ---------------------------------------------------------------------------
__global__ void merge_top2_kernel(const float4* __restrict__ part, float4* __restrict__ fin) {
    int row = blockIdx.x * blockDim.x + threadIdx.x;
    if (row >= N_PTS) return;
    float v0 = -1e30f, v1 = -1e30f; int i0 = -1;
    for (int s = 0; s < NSEG; ++s) {
        float4 p = part[(size_t)s * N_PTS + row];
        top2_ins(v0, i0, v1, p.x, __float_as_int(p.z));
        top2_ins(v0, i0, v1, p.y, -2);
    }
    fin[row] = make_float4(v0, v1, __int_as_float(i0), 0.f);
}

// ---------------------------------------------------------------------------
// Mutual-NN + ratio test. Output layout (flat, return order):
// matches (8192x2) | match_sim (8192) | mask (8192)
// ---------------------------------------------------------------------------
__global__ void finalize_kernel(const float4* __restrict__ F12, const float4* __restrict__ F21,
                                float* __restrict__ out) {
    int i = blockIdx.x * blockDim.x + threadIdx.x;
    if (i >= N_PTS) return;
    float4 f = F12[i];
    int nn12 = __float_as_int(f.z);
    float r12 = (2.f - 2.f * f.x) / ((2.f - 2.f * f.y) + K_EPS);
    int j = (nn12 < 0) ? 0 : nn12;
    float4 g = F21[j];
    int nn21 = __float_as_int(g.z);
    float r21 = (2.f - 2.f * g.x) / ((2.f - 2.f * g.y) + K_EPS);
    bool m = (nn12 >= 0) && (nn21 == i) && (r12 <= K_RATIO) && (r21 <= K_RATIO);
    out[2 * i]         = m ? (float)i    : -1.f;
    out[2 * i + 1]     = m ? (float)nn12 : -1.f;
    out[2 * N_PTS + i] = m ? f.x : 0.f;
    out[3 * N_PTS + i] = m ? 1.f : 0.f;
}

// ---------------------------------------------------------------------------
extern "C" void kernel_launch(void* const* d_in, const int* in_sizes, int n_in,
                              void* d_out, int out_size, void* d_ws, size_t ws_size,
                              hipStream_t stream) {
    (void)in_sizes; (void)n_in; (void)out_size; (void)ws_size;
    const float* A = (const float*)d_in[0];   // (704, 8192) fp32
    const float* B = (const float*)d_in[1];   // (704, 8192) fp32

    char* ws = (char*)d_ws;
    const size_t SZP = (size_t)N_PTS * D_DIM * 2;   // one bf16 plane = 11.53 MB
    __bf16* Ahi = (__bf16*)(ws);
    __bf16* Alo = (__bf16*)(ws + SZP);
    __bf16* Bhi = (__bf16*)(ws + 2 * SZP);
    __bf16* Blo = (__bf16*)(ws + 3 * SZP);
    float*  rnA = (float*)(ws + 4 * SZP);
    float*  rnB = (float*)(ws + 4 * SZP + 32768);
    float4* P12 = (float4*)(ws + 4 * SZP + 65536);
    float4* P21 = (float4*)((char*)P12 + (size_t)NSEG * N_PTS * sizeof(float4));
    float4* F12 = (float4*)((char*)P21 + (size_t)NSEG * N_PTS * sizeof(float4));
    float4* F21 = (float4*)((char*)F12 + (size_t)N_PTS * sizeof(float4));

    colnorm_kernel<<<N_PTS / 256, 256, 0, stream>>>(A, rnA);
    colnorm_kernel<<<N_PTS / 256, 256, 0, stream>>>(B, rnB);

    dim3 tg(N_PTS / 32, D_DIM / 32);
    xpose_split_kernel<<<tg, 256, 0, stream>>>(A, rnA, Ahi, Alo);
    xpose_split_kernel<<<tg, 256, 0, stream>>>(B, rnB, Bhi, Blo);

    dim3 gg(N_PTS / BM, NSEG);
    simtop2_kernel<<<gg, 256, 0, stream>>>(Ahi, Alo, Bhi, Blo, P12); // rows of d1
    simtop2_kernel<<<gg, 256, 0, stream>>>(Bhi, Blo, Ahi, Alo, P21); // rows of d2 (= cols of sim)

    merge_top2_kernel<<<N_PTS / 256, 256, 0, stream>>>(P12, F12);
    merge_top2_kernel<<<N_PTS / 256, 256, 0, stream>>>(P21, F21);

    finalize_kernel<<<N_PTS / 256, 256, 0, stream>>>(F12, F21, (float*)d_out);
}
